// LocalSelectiveSSMLayer_37245956391259
// MI455X (gfx1250) — matrix operand order repaired and checked by reading, not compiled, not run on hardware
//
#include <hip/hip_runtime.h>

// ---------------------------------------------------------------------------
// MI455X (gfx1250) implementation of LocalSelectiveSSMLayer
//   Phase A: weight/activation conversion to bf16 (transposed for WMMA-B)
//   Phase B: projections (delta/B/C) via v_wmma_f32_16x16x32_bf16
//   Phase C: sequential SSM scan, fp32 VALU, thread per (b,d), S=16 in regs
//   Phase D: router softmax + top-2 (wave32 shfl reductions)
//   Phase E: MoE gate/up GEMMs + silu, rms reduce, down GEMM with expert
//            loop fused in-tile (deterministic, no atomics)
//   Phase F: final residual rmsnorm
// GEMM K-loops are software-pipelined with ping-pong fragment buffers
// (unroll-by-2) -> partial s_wait_loadcnt and no inter-buffer v_mov copies.
// ---------------------------------------------------------------------------

typedef unsigned short u16;
typedef __attribute__((ext_vector_type(16))) __bf16 bf16x16;
typedef __attribute__((ext_vector_type(8)))  float  f32x8;

constexpr int kB = 2, kL = 2048, kD = 1024, kS = 16, kE = 4, kH = 2048;
constexpr int kN = kB * kL;           // 4096 tokens
constexpr float kEPS = 1e-6f;

// ---------------- bf16 helpers (bit-level RNE, no libdevice dependency) ----
__device__ __forceinline__ u16 f2bf(float f) {
  unsigned u = __float_as_uint(f);
  unsigned r = u + 0x7FFFu + ((u >> 16) & 1u);
  return (u16)(r >> 16);
}
__device__ __forceinline__ float bf2f(u16 h) {
  return __uint_as_float(((unsigned)h) << 16);
}

union FragU { bf16x16 v; u16 s[16]; uint4 q[2]; };

// A fragment (16x32 bf16, row-major activations): lane m = l&15, kh = l>>4
// elements [0..7] = K kh*8..+7 ; [8..15] = K 16+kh*8..+7  (two 16B chunks)
__device__ __forceinline__ bf16x16 load_fragA(const u16* __restrict__ base,
                                              size_t row, int ldk, int k0, int kh) {
  FragU f;
  const u16* p = base + row * (size_t)ldk + k0 + kh * 8;
  f.q[0] = *(const uint4*)(p);
  f.q[1] = *(const uint4*)(p + 16);
  return f.v;
}
__device__ __forceinline__ FragU load_fragA_raw(const u16* __restrict__ base,
                                                size_t row, int ldk, int k0, int kh) {
  FragU f;
  const u16* p = base + row * (size_t)ldk + k0 + kh * 8;
  f.q[0] = *(const uint4*)(p);
  f.q[1] = *(const uint4*)(p + 16);
  return f;
}
__device__ __forceinline__ bf16x16 scale_frag(FragU f, float scale) {
  FragU g;
#pragma unroll
  for (int i = 0; i < 16; ++i) g.s[i] = f2bf(bf2f(f.s[i]) * scale);
  return g.v;
}
// B fragment (32x16 bf16, weights stored transposed = column-contiguous):
// lane n = l&15, kh = l>>4; elements [0..7] = K kh*8..+7 ; [8..15] = K 16+kh*8..+7
__device__ __forceinline__ bf16x16 load_fragB(const u16* __restrict__ base,
                                              size_t row, int ldk, int k0, int kh) {
  FragU f;
  const u16* p = base + row * (size_t)ldk + k0 + kh * 8;
  f.q[0] = *(const uint4*)(p);
  f.q[1] = *(const uint4*)(p + 16);
  return f.v;
}

__device__ __forceinline__ f32x8 wmma_bf16(bf16x16 a, bf16x16 b, f32x8 c) {
  return __builtin_amdgcn_wmma_f32_16x16x32_bf16(false, a, false, b,
                                                 (short)0, c, false, false);
}

// ---------------- Phase A: conversion kernels ------------------------------
__global__ __launch_bounds__(256) void k_f32_to_bf16(const float* __restrict__ src,
                                                     u16* __restrict__ dst, long n) {
  for (long i = blockIdx.x * (long)blockDim.x + threadIdx.x; i < n;
       i += (long)gridDim.x * blockDim.x)
    dst[i] = f2bf(src[i]);
}

// dst[c*R + r] = bf16(src[r*C + c] * rowscale[r])   (per blockIdx.z slice)
__global__ __launch_bounds__(256) void k_transpose_bf16(
    const float* __restrict__ src, u16* __restrict__ dst, int R, int C,
    const float* __restrict__ rowscale, long srcZ, long dstZ, long sclZ) {
  long z = blockIdx.z;
  const float* s = src + z * srcZ;
  u16* d = dst + z * dstZ;
  const float* rs = rowscale ? rowscale + z * sclZ : nullptr;
  long total = (long)R * C;
  for (long i = blockIdx.x * (long)blockDim.x + threadIdx.x; i < total;
       i += (long)gridDim.x * blockDim.x) {
    long r = i / C, c = i - r * C;
    float v = s[r * (long)C + c];
    if (rs) v *= rs[r];
    d[c * (long)R + r] = f2bf(v);
  }
}

// ---------------- Phase B: projections -------------------------------------
// delta = softplus(x @ W_delta + b); wave computes 16x64 tile, K=1024
__global__ __launch_bounds__(256) void k_proj_delta(
    const u16* __restrict__ A, const u16* __restrict__ Bt,
    const float* __restrict__ bias, float* __restrict__ out) {
  int gwid = (blockIdx.x * blockDim.x + threadIdx.x) >> 5;
  int lane = threadIdx.x & 31;
  constexpr int tilesN = kD / 64;                 // 16
  if (gwid >= (kN / 16) * tilesN) return;
  int m0 = (gwid / tilesN) * 16, n0 = (gwid % tilesN) * 64;
  int mr = lane & 15, kh = lane >> 4;
  size_t arow = m0 + mr;
  f32x8 acc[4] = {};
  // ping-pong software pipeline (no inter-buffer copies)
  bf16x16 a0, a1, b0[4], b1[4];
  a0 = load_fragA(A, arow, kD, 0, kh);
#pragma unroll
  for (int j = 0; j < 4; ++j) b0[j] = load_fragB(Bt, n0 + j * 16 + mr, kD, 0, kh);
  int k0 = 0;
  for (; k0 + 64 < kD; k0 += 64) {
    __builtin_prefetch(A + arow * kD + k0 + 64, 0, 1);
    a1 = load_fragA(A, arow, kD, k0 + 32, kh);
#pragma unroll
    for (int j = 0; j < 4; ++j) b1[j] = load_fragB(Bt, n0 + j * 16 + mr, kD, k0 + 32, kh);
#pragma unroll
    for (int j = 0; j < 4; ++j) acc[j] = wmma_bf16(a0, b0[j], acc[j]);
    a0 = load_fragA(A, arow, kD, k0 + 64, kh);
#pragma unroll
    for (int j = 0; j < 4; ++j) b0[j] = load_fragB(Bt, n0 + j * 16 + mr, kD, k0 + 64, kh);
#pragma unroll
    for (int j = 0; j < 4; ++j) acc[j] = wmma_bf16(a1, b1[j], acc[j]);
  }
  a1 = load_fragA(A, arow, kD, k0 + 32, kh);
#pragma unroll
  for (int j = 0; j < 4; ++j) b1[j] = load_fragB(Bt, n0 + j * 16 + mr, kD, k0 + 32, kh);
#pragma unroll
  for (int j = 0; j < 4; ++j) acc[j] = wmma_bf16(a0, b0[j], acc[j]);
#pragma unroll
  for (int j = 0; j < 4; ++j) acc[j] = wmma_bf16(a1, b1[j], acc[j]);

  int nL = lane & 15, mB = (lane >> 4) * 8;
#pragma unroll
  for (int j = 0; j < 4; ++j)
#pragma unroll
    for (int r = 0; r < 8; ++r) {
      int row = m0 + mB + r, col = n0 + j * 16 + nL;
      float v = acc[j][r] + bias[col];
      out[(size_t)row * kD + col] = (v > 20.f) ? v : log1pf(__expf(v));
    }
}

// Bm = x @ W_B, Cm = x @ W_C  (N x 16 each, single 16-wide WMMA tile)
__global__ __launch_bounds__(256) void k_proj_bc(
    const u16* __restrict__ A, const u16* __restrict__ WBT,
    const u16* __restrict__ WCT, float* __restrict__ Bm, float* __restrict__ Cm) {
  int gwid = (blockIdx.x * blockDim.x + threadIdx.x) >> 5;
  int lane = threadIdx.x & 31;
  int which = gwid >> 8;                          // kN/16 == 256 tiles each
  int tm = gwid & 255;
  if (which > 1) return;
  const u16* Bt = which ? WCT : WBT;
  float* outp = which ? Cm : Bm;
  int m0 = tm * 16, mr = lane & 15, kh = lane >> 4;
  size_t arow = m0 + mr;
  f32x8 acc = {};
  bf16x16 a0 = load_fragA(A, arow, kD, 0, kh);
  bf16x16 c0 = load_fragB(Bt, mr, kD, 0, kh);
  bf16x16 a1, c1;
  int k0 = 0;
  for (; k0 + 64 < kD; k0 += 64) {
    a1 = load_fragA(A, arow, kD, k0 + 32, kh);
    c1 = load_fragB(Bt, mr, kD, k0 + 32, kh);
    acc = wmma_bf16(a0, c0, acc);
    a0 = load_fragA(A, arow, kD, k0 + 64, kh);
    c0 = load_fragB(Bt, mr, kD, k0 + 64, kh);
    acc = wmma_bf16(a1, c1, acc);
  }
  a1 = load_fragA(A, arow, kD, k0 + 32, kh);
  c1 = load_fragB(Bt, mr, kD, k0 + 32, kh);
  acc = wmma_bf16(a0, c0, acc);
  acc = wmma_bf16(a1, c1, acc);
  int nL = lane & 15, mB = (lane >> 4) * 8;
#pragma unroll
  for (int r = 0; r < 8; ++r) outp[(size_t)(m0 + mB + r) * kS + nL] = acc[r];
}

// ---------------- Phase C: sequential SSM scan -----------------------------
__global__ __launch_bounds__(256) void k_scan(
    const float* __restrict__ x, const float* __restrict__ delta,
    const float* __restrict__ Bm, const float* __restrict__ Cm,
    const float* __restrict__ A_log, const float* __restrict__ Dp,
    float* __restrict__ ssmF, u16* __restrict__ ssmBF) {
  int tid = blockIdx.x * blockDim.x + threadIdx.x;     // kB*kD threads
  if (tid >= kB * kD) return;
  int b = tid >> 10, d = tid & (kD - 1);
  float Arow[kS], h[kS];
#pragma unroll
  for (int s = 0; s < kS; ++s) { Arow[s] = -__expf(A_log[d * kS + s]); h[s] = 0.f; }
  float dpar = Dp[d];
  for (int t = 0; t < kL; ++t) {
    size_t tok = (size_t)b * kL + t;
    float xt = x[tok * kD + d];
    float dl = delta[tok * kD + d];
    const float* Bt = Bm + tok * kS;
    const float* Ct = Cm + tok * kS;
    float y = 0.f;
#pragma unroll
    for (int s = 0; s < kS; ++s) {
      float barA = __expf(fminf(dl * Arow[s], 10.f));
      float barB = fminf(fmaxf(dl * Bt[s], -10.f), 10.f);
      float hn = barA * h[s] + barB * xt;
      hn = fminf(fmaxf(hn, -10000.f), 10000.f);
      h[s] = hn;
      y += hn * Ct[s];
    }
    y += xt * dpar;
    ssmF[tok * kD + d] = y;
    ssmBF[tok * kD + d] = f2bf(y);
  }
}

// ---------------- Phase D: router + top-2 gates ----------------------------
__global__ __launch_bounds__(256) void k_router(const float* __restrict__ ssm,
                                                const float* __restrict__ Wr,
                                                float* __restrict__ gate) {
  int gwid = (blockIdx.x * blockDim.x + threadIdx.x) >> 5;
  int lane = threadIdx.x & 31;
  if (gwid >= kN) return;
  const float* row = ssm + (size_t)gwid * kD;
  const float4* Wr4 = (const float4*)Wr;               // E=4 contiguous per d
  float l0 = 0.f, l1 = 0.f, l2 = 0.f, l3 = 0.f;
#pragma unroll 4
  for (int i = 0; i < kD / 32; ++i) {
    int dI = lane + i * 32;
    float v = row[dI];
    float4 w = Wr4[dI];
    l0 += v * w.x; l1 += v * w.y; l2 += v * w.z; l3 += v * w.w;
  }
#pragma unroll
  for (int off = 16; off > 0; off >>= 1) {
    l0 += __shfl_xor(l0, off, 32); l1 += __shfl_xor(l1, off, 32);
    l2 += __shfl_xor(l2, off, 32); l3 += __shfl_xor(l3, off, 32);
  }
  if (lane == 0) {
    float m = fmaxf(fmaxf(l0, l1), fmaxf(l2, l3));
    float p[4] = {__expf(l0 - m), __expf(l1 - m), __expf(l2 - m), __expf(l3 - m)};
    float sum = p[0] + p[1] + p[2] + p[3];
#pragma unroll
    for (int e = 0; e < 4; ++e) p[e] /= sum;
    int i0 = 0;
#pragma unroll
    for (int e = 1; e < 4; ++e) if (p[e] > p[i0]) i0 = e;
    int i1 = (i0 == 0) ? 1 : 0;
#pragma unroll
    for (int e = 0; e < 4; ++e) if (e != i0 && p[e] > p[i1]) i1 = e;
    float inv = 1.f / (p[i0] + p[i1] + 1e-9f);
    float w0 = p[i0] * inv, w1 = p[i1] * inv;
#pragma unroll
    for (int e = 0; e < 4; ++e)
      gate[(size_t)e * kN + gwid] = (e == i0) ? w0 : ((e == i1) ? w1 : 0.f);
  }
}

// ---------------- Phase E1: gate/up GEMMs + silu ---------------------------
// 8 accumulators -> no cross-k double buffer; instead batch the whole
// k-step's 18 loads before the 8 WMMAs so waits can be partial.
__global__ __launch_bounds__(256) void k_moe_gu(
    const u16* __restrict__ Abf, const u16* __restrict__ WgT,
    const u16* __restrict__ WuT, u16* __restrict__ hact) {
  int e = blockIdx.z;
  int gwid = (blockIdx.x * blockDim.x + threadIdx.x) >> 5;
  int lane = threadIdx.x & 31;
  constexpr int tilesN = kH / 64;                 // 32
  if (gwid >= (kN / 16) * tilesN) return;
  int m0 = (gwid / tilesN) * 16, n0 = (gwid % tilesN) * 64;
  const u16* Bg = WgT + (size_t)e * kH * kD;
  const u16* Bu = WuT + (size_t)e * kH * kD;
  u16* dst = hact + (size_t)e * kN * kH;
  int mr = lane & 15, kh = lane >> 4;
  f32x8 ag[4] = {}, au[4] = {};
  for (int k0 = 0; k0 < kD; k0 += 32) {
    if (k0 + 32 < kD)
      __builtin_prefetch(Abf + (size_t)(m0 + mr) * kD + k0 + 32, 0, 1);
    bf16x16 a = load_fragA(Abf, m0 + mr, kD, k0, kh);
    bf16x16 bg[4], bu[4];
#pragma unroll
    for (int j = 0; j < 4; ++j) bg[j] = load_fragB(Bg, n0 + j * 16 + mr, kD, k0, kh);
#pragma unroll
    for (int j = 0; j < 4; ++j) bu[j] = load_fragB(Bu, n0 + j * 16 + mr, kD, k0, kh);
#pragma unroll
    for (int j = 0; j < 4; ++j) {
      ag[j] = wmma_bf16(a, bg[j], ag[j]);
      au[j] = wmma_bf16(a, bu[j], au[j]);
    }
  }
  int nL = lane & 15, mB = (lane >> 4) * 8;
#pragma unroll
  for (int j = 0; j < 4; ++j)
#pragma unroll
    for (int r = 0; r < 8; ++r) {
      int row = m0 + mB + r, col = n0 + j * 16 + nL;
      float g = ag[j][r];
      float v = g / (1.f + __expf(-g)) * au[j][r];       // silu(g)*u
      dst[(size_t)row * kH + col] = f2bf(v);
    }
}

// ---------------- Phase E2: rms over H per (expert, token) -----------------
__global__ __launch_bounds__(256) void k_rms(const u16* __restrict__ hact,
                                             float* __restrict__ invrms) {
  int gwid = (blockIdx.x * blockDim.x + threadIdx.x) >> 5;
  int lane = threadIdx.x & 31;
  if (gwid >= kE * kN) return;
  const u16* row = hact + (size_t)gwid * kH;
  float ss = 0.f;
#pragma unroll 4
  for (int i = 0; i < kH / 32; ++i) {
    float v = bf2f(row[lane + i * 32]);
    ss += v * v;
  }
#pragma unroll
  for (int off = 16; off > 0; off >>= 1) ss += __shfl_xor(ss, off, 32);
  if (lane == 0) invrms[gwid] = rsqrtf(ss / (float)kH + kEPS);
}

// ---------------- Phase E3: down GEMM, expert loop fused in-tile -----------
__global__ __launch_bounds__(256) void k_moe_down(
    const u16* __restrict__ hact, const u16* __restrict__ WdT,
    const float* __restrict__ invrms, const float* __restrict__ gate,
    float* __restrict__ moe) {
  int gwid = (blockIdx.x * blockDim.x + threadIdx.x) >> 5;
  int lane = threadIdx.x & 31;
  constexpr int tilesN = kD / 64;                 // 16
  if (gwid >= (kN / 16) * tilesN) return;
  int m0 = (gwid / tilesN) * 16, n0 = (gwid % tilesN) * 64;
  int mr = lane & 15, kh = lane >> 4;
  int mB = (lane >> 4) * 8, nL = lane & 15;
  f32x8 accF[4] = {};
  for (int e = 0; e < kE; ++e) {
    const u16* Ae = hact + (size_t)e * kN * kH;
    const u16* Be = WdT + (size_t)e * kD * kH;
    float irs = invrms[(size_t)e * kN + m0 + mr];        // lane's token row
    size_t arow = m0 + mr;
    f32x8 acc[4] = {};
    // ping-pong pipeline; invrms scaling at use time overlaps in-flight loads
    FragU ar0 = load_fragA_raw(Ae, arow, kH, 0, kh);
    bf16x16 b0[4], b1[4];
    FragU ar1;
#pragma unroll
    for (int j = 0; j < 4; ++j) b0[j] = load_fragB(Be, n0 + j * 16 + mr, kH, 0, kh);
    int k0 = 0;
    for (; k0 + 64 < kH; k0 += 64) {
      __builtin_prefetch(Ae + arow * kH + k0 + 64, 0, 1);
      ar1 = load_fragA_raw(Ae, arow, kH, k0 + 32, kh);
#pragma unroll
      for (int j = 0; j < 4; ++j) b1[j] = load_fragB(Be, n0 + j * 16 + mr, kH, k0 + 32, kh);
      {
        bf16x16 a = scale_frag(ar0, irs);
#pragma unroll
        for (int j = 0; j < 4; ++j) acc[j] = wmma_bf16(a, b0[j], acc[j]);
      }
      ar0 = load_fragA_raw(Ae, arow, kH, k0 + 64, kh);
#pragma unroll
      for (int j = 0; j < 4; ++j) b0[j] = load_fragB(Be, n0 + j * 16 + mr, kH, k0 + 64, kh);
      {
        bf16x16 a = scale_frag(ar1, irs);
#pragma unroll
        for (int j = 0; j < 4; ++j) acc[j] = wmma_bf16(a, b1[j], acc[j]);
      }
    }
    ar1 = load_fragA_raw(Ae, arow, kH, k0 + 32, kh);
#pragma unroll
    for (int j = 0; j < 4; ++j) b1[j] = load_fragB(Be, n0 + j * 16 + mr, kH, k0 + 32, kh);
    {
      bf16x16 a = scale_frag(ar0, irs);
#pragma unroll
      for (int j = 0; j < 4; ++j) acc[j] = wmma_bf16(a, b0[j], acc[j]);
    }
    {
      bf16x16 a = scale_frag(ar1, irs);
#pragma unroll
      for (int j = 0; j < 4; ++j) acc[j] = wmma_bf16(a, b1[j], acc[j]);
    }

    float gr[8];
#pragma unroll
    for (int r = 0; r < 8; ++r) gr[r] = gate[(size_t)e * kN + m0 + mB + r];
#pragma unroll
    for (int j = 0; j < 4; ++j)
#pragma unroll
      for (int r = 0; r < 8; ++r) accF[j][r] += gr[r] * acc[j][r];
  }
#pragma unroll
  for (int j = 0; j < 4; ++j)
#pragma unroll
    for (int r = 0; r < 8; ++r)
      moe[(size_t)(m0 + mB + r) * kD + (n0 + j * 16 + nL)] = accF[j][r];
}

// ---------------- Phase F: final residual rmsnorm --------------------------
__global__ __launch_bounds__(256) void k_final(const float* __restrict__ ssm,
                                               const float* __restrict__ moe,
                                               const float* __restrict__ nw,
                                               float* __restrict__ out) {
  int gwid = (blockIdx.x * blockDim.x + threadIdx.x) >> 5;
  int lane = threadIdx.x & 31;
  if (gwid >= kN) return;
  size_t base = (size_t)gwid * kD;
  float vals[kD / 32];
  float ss = 0.f;
#pragma unroll 4
  for (int i = 0; i < kD / 32; ++i) {
    int dI = lane + i * 32;
    float v = ssm[base + dI] + moe[base + dI];
    vals[i] = v;
    ss += v * v;
  }
#pragma unroll
  for (int off = 16; off > 0; off >>= 1) ss += __shfl_xor(ss, off, 32);
  float ir = rsqrtf(ss / (float)kD + kEPS);
#pragma unroll 4
  for (int i = 0; i < kD / 32; ++i) {
    int dI = lane + i * 32;
    out[base + dI] = nw[dI] * vals[i] * ir;
  }
}

// ---------------------------------------------------------------------------
extern "C" void kernel_launch(void* const* d_in, const int* in_sizes, int n_in,
                              void* d_out, int out_size, void* d_ws, size_t ws_size,
                              hipStream_t stream) {
  (void)in_sizes; (void)n_in; (void)out_size; (void)ws_size;
  const float* x       = (const float*)d_in[0];
  const float* A_log   = (const float*)d_in[1];
  const float* D_param = (const float*)d_in[2];
  const float* W_delta = (const float*)d_in[3];
  const float* b_delta = (const float*)d_in[4];
  const float* W_B     = (const float*)d_in[5];
  const float* W_C     = (const float*)d_in[6];
  const float* W_router= (const float*)d_in[7];
  const float* Wg      = (const float*)d_in[8];
  const float* Wu      = (const float*)d_in[9];
  const float* Wd      = (const float*)d_in[10];
  const float* wn_exp  = (const float*)d_in[11];
  const float* norm_w  = (const float*)d_in[12];
  float* out = (float*)d_out;

  unsigned char* w = (unsigned char*)d_ws;
  size_t off = 0;
  auto take = [&](size_t bytes) -> void* {
    void* p = w + off;
    off += (bytes + 255) & ~(size_t)255;
    return p;
  };
  u16*   xbf    = (u16*)  take((size_t)kN * kD * 2);
  u16*   WdelT  = (u16*)  take((size_t)kD * kD * 2);
  u16*   WBT    = (u16*)  take((size_t)kS * kD * 2);
  u16*   WCT    = (u16*)  take((size_t)kS * kD * 2);
  u16*   WgT    = (u16*)  take((size_t)kE * kH * kD * 2);
  u16*   WuT    = (u16*)  take((size_t)kE * kH * kD * 2);
  u16*   WdTs   = (u16*)  take((size_t)kE * kD * kH * 2);
  float* delta  = (float*)take((size_t)kN * kD * 4);
  float* Bm     = (float*)take((size_t)kN * kS * 4);
  float* Cm     = (float*)take((size_t)kN * kS * 4);
  float* ssmF   = (float*)take((size_t)kN * kD * 4);
  u16*   ssmBF  = (u16*)  take((size_t)kN * kD * 2);
  float* gate   = (float*)take((size_t)kE * kN * 4);
  u16*   hact   = (u16*)  take((size_t)kE * kN * kH * 2);
  float* invrms = (float*)take((size_t)kE * kN * 4);
  float* moe    = (float*)take((size_t)kN * kD * 4);

  // Phase A: conversions / transposes (one HBM pass; bf16 weights fit in L2)
  k_f32_to_bf16<<<2048, 256, 0, stream>>>(x, xbf, (long)kN * kD);
  k_transpose_bf16<<<dim3(512, 1, 1), 256, 0, stream>>>(W_delta, WdelT, kD, kD,
                                                        nullptr, 0, 0, 0);
  k_transpose_bf16<<<dim3(64, 1, 1), 256, 0, stream>>>(W_B, WBT, kD, kS,
                                                       nullptr, 0, 0, 0);
  k_transpose_bf16<<<dim3(64, 1, 1), 256, 0, stream>>>(W_C, WCT, kD, kS,
                                                       nullptr, 0, 0, 0);
  k_transpose_bf16<<<dim3(512, 1, kE), 256, 0, stream>>>(
      Wg, WgT, kD, kH, nullptr, (long)kD * kH, (long)kH * kD, 0);
  k_transpose_bf16<<<dim3(512, 1, kE), 256, 0, stream>>>(
      Wu, WuT, kD, kH, nullptr, (long)kD * kH, (long)kH * kD, 0);
  // fold rmsnorm weight wn_exp[e][h] into Wd rows during transpose
  k_transpose_bf16<<<dim3(512, 1, kE), 256, 0, stream>>>(
      Wd, WdTs, kH, kD, wn_exp, (long)kH * kD, (long)kD * kH, (long)kH);

  // Phase B: projections (WMMA)
  k_proj_delta<<<(kN / 16) * (kD / 64) / 8, 256, 0, stream>>>(xbf, WdelT,
                                                              b_delta, delta);
  k_proj_bc<<<(2 * (kN / 16)) / 8, 256, 0, stream>>>(xbf, WBT, WCT, Bm, Cm);

  // Phase C: sequential scan
  k_scan<<<(kB * kD) / 256, 256, 0, stream>>>(x, delta, Bm, Cm, A_log, D_param,
                                              ssmF, ssmBF);

  // Phase D: router
  k_router<<<kN / 8, 256, 0, stream>>>(ssmF, W_router, gate);

  // Phase E: MoE
  k_moe_gu<<<dim3((kN / 16) * (kH / 64) / 8, 1, kE), 256, 0, stream>>>(
      ssmBF, WgT, WuT, hact);
  k_rms<<<(kE * kN) / 8, 256, 0, stream>>>(hact, invrms);
  k_moe_down<<<(kN / 16) * (kD / 64) / 8, 256, 0, stream>>>(hact, WdTs, invrms,
                                                            gate, moe);

  // Phase F: final rmsnorm
  k_final<<<kN / 8, 256, 0, stream>>>(ssmF, moe, norm_w, out);
}
